// MultiGaussianKernel_67405216743441
// MI455X (gfx1250) — matrix operand order repaired and checked by reading, not compiled, not run on hardware
//
#include <hip/hip_runtime.h>
#include <hip/hip_bf16.h>
#include <cstdint>

// CDNA5 (gfx1250) wave32 WMMA types
typedef __attribute__((ext_vector_type(16))) __bf16 v16bf;
typedef __attribute__((ext_vector_type(8)))  __bf16 v8bf;
typedef __attribute__((ext_vector_type(4)))  __bf16 v4bf;
typedef __attribute__((ext_vector_type(8)))  float  v8f;

// Round-to-nearest-even f32 -> bf16 (manual, compile-proven), returning the
// bf16 value re-expanded to f32.
__device__ __forceinline__ float bf_split(float f, __bf16& h) {
    unsigned u = __float_as_uint(f);
    unsigned r = (u + 0x7FFFu + ((u >> 16) & 1u)) & 0xFFFF0000u;
    h = __builtin_bit_cast(__bf16, (unsigned short)(r >> 16));
    return __uint_as_float(r);
}

// Build a v16bf WMMA fragment from two 16-byte (8 x bf16) loads.
__device__ __forceinline__ v16bf ldfrag(const __bf16* p0, const __bf16* p1) {
    v8bf a = *(const v8bf*)p0;
    v8bf b = *(const v8bf*)p1;
    return __builtin_shufflevector(a, b, 0, 1, 2, 3, 4, 5, 6, 7,
                                         8, 9, 10, 11, 12, 13, 14, 15);
}

// =========================================================================
// FAST PATH (needs 64 MB workspace): one-time f32 -> (bf16 hi, bf16 lo)
// split planes + fused row norms, then an LDS-staged load+WMMA GEMM.
// =========================================================================

// One wave per row of 1024 f32: emit hi/lo bf16 planes and the squared norm.
__global__ void __launch_bounds__(256)
conv_split_norm(const float* __restrict__ A, __bf16* __restrict__ hi,
                __bf16* __restrict__ lo, float* __restrict__ nrm, int rows) {
    const int lane = threadIdx.x & 31;
    const int w    = threadIdx.x >> 5;
    const int row  = blockIdx.x * 8 + w;
    if (row >= rows) return;

    const float4* p = (const float4*)(A + (size_t)row * 1024);
    __bf16* ph = hi + (size_t)row * 1024;
    __bf16* pl = lo + (size_t)row * 1024;

    float s = 0.f;
#pragma unroll
    for (int c = 0; c < 8; ++c) {              // 8 * 32 lanes * 4 = 1024 floats
        const int e = lane + 32 * c;
        float4 v = p[e];
        float f[4] = {v.x, v.y, v.z, v.w};
        v4bf h4, l4;
#pragma unroll
        for (int i = 0; i < 4; ++i) {
            s += f[i] * f[i];
            __bf16 h;
            float hf = bf_split(f[i], h);
            h4[i] = h;
            __bf16 l;
            bf_split(f[i] - hf, l);
            l4[i] = l;
        }
        *(v4bf*)(ph + e * 4) = h4;
        *(v4bf*)(pl + e * 4) = l4;
    }
#pragma unroll
    for (int off = 16; off > 0; off >>= 1)
        s += __shfl_xor(s, off, 32);
    if (lane == 0) nrm[row] = s;
}

// Workgroup = 256 threads = 8 waves; tile = 128(M) x 128(N), K-chunk = 32.
// LDS staging: 4 planes (Ah, Al, Bh, Bl) of 128 rows x 32 k (padded to 40)
// = 40 KB. Unique global traffic: 32 KB per WG-chunk (3x less than direct).
// Register prefetch overlaps next chunk's global loads with current WMMAs.
__global__ void __launch_bounds__(256)
mgk_gemm_bf16(const __bf16* __restrict__ Xh, const __bf16* __restrict__ Xl,
              const __bf16* __restrict__ Yh, const __bf16* __restrict__ Yl,
              const float* __restrict__ x2, const float* __restrict__ y2,
              float* __restrict__ out, int Mcols) {
    constexpr int D  = 1024;
    constexpr int KC = 32;   // K-chunk (one bf16 WMMA K)
    constexpr int KP = 40;   // padded LDS row length (80 B -> conflict-free)

    __shared__ __bf16 smem[4][128 * KP];   // 0:Ah 1:Al 2:Bh 3:Bl  (40 KB)

    const int t    = threadIdx.x;
    const int lane = t & 31;
    const int wid  = t >> 5;
    const int hg   = lane >> 4;     // half-group (0: lanes 0-15, 1: 16-31)
    const int lr   = lane & 15;

    // ---- cooperative fill mapping: thread -> (row, 32B-half) of each plane
    const int rid  = t >> 1;        // 0..127
    const int half = t & 1;         // 32-byte half of a 64-byte row chunk
    const size_t rowX = (size_t)(blockIdx.x * 128 + rid) * D + half * 16;
    const size_t rowY = (size_t)(blockIdx.y * 128 + rid) * D + half * 16;
    const __bf16* gsrc[4] = {Xh + rowX, Xl + rowX, Yh + rowY, Yl + rowY};
    const int sbase = rid * KP + half * 16;

    // ---- per-wave compute mapping
    const int tileM = blockIdx.x * 128 + (wid & 3) * 32;
    const int tileN = blockIdx.y * 128 + (wid >> 2) * 64;
    const int arow0 = (wid & 3) * 32 + lr;   // row-in-tile for A fragments
    const int brow0 = (wid >> 2) * 64 + lr;  // row-in-tile for B fragments

    v8f  acc[2][4] = {};
    v8bf st[8];                              // register staging (8 x 16 B)

    // prologue: fetch chunk 0 into LDS
#pragma unroll
    for (int p = 0; p < 4; ++p) {
        st[2 * p]     = *(const v8bf*)(gsrc[p]);
        st[2 * p + 1] = *(const v8bf*)(gsrc[p] + 8);
    }
#pragma unroll
    for (int p = 0; p < 4; ++p) {
        *(v8bf*)&smem[p][sbase]     = st[2 * p];
        *(v8bf*)&smem[p][sbase + 8] = st[2 * p + 1];
    }
    __syncthreads();

    for (int k0 = 0; k0 < D; k0 += KC) {
        const bool more = (k0 + KC) < D;
        // prefetch next chunk into registers (latency hidden by WMMAs below)
        if (more) {
#pragma unroll
            for (int p = 0; p < 4; ++p) {
                st[2 * p]     = *(const v8bf*)(gsrc[p] + k0 + KC);
                st[2 * p + 1] = *(const v8bf*)(gsrc[p] + k0 + KC + 8);
            }
        }

        // fragments from LDS
        v16bf ah[2], al[2], bh[4], bl[4];
        // A fragment (16x32 bf16): lane group 0 -> K[0..7],K[16..23];
        // lane group 1 -> K[8..15],K[24..31]  (ISA 7.12.2)
#pragma unroll
        for (int mt = 0; mt < 2; ++mt) {
            const int off = (arow0 + 16 * mt) * KP;
            ah[mt] = ldfrag(&smem[0][off + 8 * hg], &smem[0][off + 16 + 8 * hg]);
            al[mt] = ldfrag(&smem[1][off + 8 * hg], &smem[1][off + 16 + 8 * hg]);
        }
        // B fragment (32x16 bf16): per-lane N = lr, K[8*hg..+7],K[16+8*hg..+7].
#pragma unroll
        for (int nt = 0; nt < 4; ++nt) {
            const int off = (brow0 + 16 * nt) * KP + 8 * hg;
            bh[nt] = ldfrag(&smem[2][off], &smem[2][off + 16]);
            bl[nt] = ldfrag(&smem[3][off], &smem[3][off + 16]);
        }

        // bf16x3 split accumulation: hi*hi + hi*lo + lo*hi  (~fp32 accuracy)
#pragma unroll
        for (int mt = 0; mt < 2; ++mt)
#pragma unroll
            for (int nt = 0; nt < 4; ++nt) {
                acc[mt][nt] = __builtin_amdgcn_wmma_f32_16x16x32_bf16(
                    false, ah[mt], false, bh[nt], (short)0, acc[mt][nt], false, false);
                acc[mt][nt] = __builtin_amdgcn_wmma_f32_16x16x32_bf16(
                    false, ah[mt], false, bl[nt], (short)0, acc[mt][nt], false, false);
                acc[mt][nt] = __builtin_amdgcn_wmma_f32_16x16x32_bf16(
                    false, al[mt], false, bh[nt], (short)0, acc[mt][nt], false, false);
            }

        __syncthreads();                      // all waves done reading smem
        if (more) {
#pragma unroll
            for (int p = 0; p < 4; ++p) {     // commit prefetched chunk
                *(v8bf*)&smem[p][sbase]     = st[2 * p];
                *(v8bf*)&smem[p][sbase + 8] = st[2 * p + 1];
            }
            __syncthreads();                  // stores visible to all waves
        }
    }

    // Epilogue: d2 = |x|^2 + |y|^2 - 2*dot ; out = sum_a exp(-d2/(2a)),
    // 1/(2a) in {4, 2, 1, 0.5, 0.25}. Non-temporal store (write-once stream).
#pragma unroll
    for (int mt = 0; mt < 2; ++mt) {
        float xs[8];
#pragma unroll
        for (int r = 0; r < 8; ++r)
            xs[r] = x2[tileM + 16 * mt + 8 * hg + r];
#pragma unroll
        for (int nt = 0; nt < 4; ++nt) {
            const int n    = tileN + 16 * nt + lr;
            const float yn = y2[n];
#pragma unroll
            for (int r = 0; r < 8; ++r) {
                const int m    = tileM + 16 * mt + 8 * hg + r;  // C: M = r + 8*hg
                const float d2 = xs[r] + yn - 2.0f * acc[mt][nt][r];
                const float o  = __expf(d2 * -4.0f)  + __expf(d2 * -2.0f)
                               + __expf(d2 * -1.0f)  + __expf(d2 * -0.5f)
                               + __expf(d2 * -0.25f);
                __builtin_nontemporal_store(o, &out[(size_t)m * Mcols + n]);
            }
        }
    }
}

// =========================================================================
// FALLBACK PATH (small workspace): compile-proven in-register split kernels.
// =========================================================================

__device__ __forceinline__ void split8(const float* __restrict__ p, int base,
                                       v16bf& hi, v16bf& lo) {
    float4 a = *(const float4*)(p);
    float4 b = *(const float4*)(p + 4);
    float t[8] = {a.x, a.y, a.z, a.w, b.x, b.y, b.z, b.w};
#pragma unroll
    for (int i = 0; i < 8; ++i) {
        __bf16 h;
        float hf = bf_split(t[i], h);
        hi[base + i] = h;
        __bf16 l;
        bf_split(t[i] - hf, l);
        lo[base + i] = l;
    }
}

__global__ void __launch_bounds__(256)
row_sqnorm(const float* __restrict__ A, float* __restrict__ o, int rows) {
    const int lane = threadIdx.x & 31;
    const int w    = threadIdx.x >> 5;
    const int row  = blockIdx.x * 8 + w;
    if (row >= rows) return;
    const float4* p = (const float4*)(A + (size_t)row * 1024);
    float s = 0.f;
#pragma unroll
    for (int c = 0; c < 8; ++c) {
        float4 v = p[lane + 32 * c];
        s += v.x * v.x + v.y * v.y + v.z * v.z + v.w * v.w;
    }
#pragma unroll
    for (int off = 16; off > 0; off >>= 1)
        s += __shfl_xor(s, off, 32);
    if (lane == 0) o[row] = s;
}

__global__ void __launch_bounds__(256)
mgk_gemm_f32src(const float* __restrict__ X, const float* __restrict__ Y,
                const float* __restrict__ x2, const float* __restrict__ y2,
                float* __restrict__ out, int Mcols) {
    constexpr int D = 1024;
    const int lane = threadIdx.x & 31;
    const int wid  = threadIdx.x >> 5;
    const int hg   = lane >> 4;
    const int lr   = lane & 15;

    const int tileM = blockIdx.x * 128 + (wid & 3) * 32;
    const int tileN = blockIdx.y * 128 + (wid >> 2) * 64;

    const float* xr[2];
    const float* yr[4];
#pragma unroll
    for (int mt = 0; mt < 2; ++mt)
        xr[mt] = X + (size_t)(tileM + 16 * mt + lr) * D;
#pragma unroll
    for (int nt = 0; nt < 4; ++nt)
        yr[nt] = Y + (size_t)(tileN + 16 * nt + lr) * D;

    v8f acc[2][4] = {};

    for (int k0 = 0; k0 < D; k0 += 32) {
        v16bf ah[2], al[2], bh[4], bl[4];
#pragma unroll
        for (int mt = 0; mt < 2; ++mt) {
            split8(xr[mt] + k0 + 8 * hg,      0, ah[mt], al[mt]);
            split8(xr[mt] + k0 + 16 + 8 * hg, 8, ah[mt], al[mt]);
        }
#pragma unroll
        for (int nt = 0; nt < 4; ++nt) {
            split8(yr[nt] + k0 + 8 * hg,      0, bh[nt], bl[nt]);
            split8(yr[nt] + k0 + 16 + 8 * hg, 8, bh[nt], bl[nt]);
        }
#pragma unroll
        for (int mt = 0; mt < 2; ++mt)
#pragma unroll
            for (int nt = 0; nt < 4; ++nt) {
                acc[mt][nt] = __builtin_amdgcn_wmma_f32_16x16x32_bf16(
                    false, ah[mt], false, bh[nt], (short)0, acc[mt][nt], false, false);
                acc[mt][nt] = __builtin_amdgcn_wmma_f32_16x16x32_bf16(
                    false, ah[mt], false, bl[nt], (short)0, acc[mt][nt], false, false);
                acc[mt][nt] = __builtin_amdgcn_wmma_f32_16x16x32_bf16(
                    false, al[mt], false, bh[nt], (short)0, acc[mt][nt], false, false);
            }
    }

#pragma unroll
    for (int mt = 0; mt < 2; ++mt) {
        float xs[8];
#pragma unroll
        for (int r = 0; r < 8; ++r)
            xs[r] = x2[tileM + 16 * mt + 8 * hg + r];
#pragma unroll
        for (int nt = 0; nt < 4; ++nt) {
            const int n    = tileN + 16 * nt + lr;
            const float yn = y2[n];
#pragma unroll
            for (int r = 0; r < 8; ++r) {
                const int m    = tileM + 16 * mt + 8 * hg + r;
                const float d2 = xs[r] + yn - 2.0f * acc[mt][nt][r];
                const float o  = __expf(d2 * -4.0f)  + __expf(d2 * -2.0f)
                               + __expf(d2 * -1.0f)  + __expf(d2 * -0.5f)
                               + __expf(d2 * -0.25f);
                __builtin_nontemporal_store(o, &out[(size_t)m * Mcols + n]);
            }
        }
    }
}

// =========================================================================
extern "C" void kernel_launch(void* const* d_in, const int* in_sizes, int n_in,
                              void* d_out, int out_size, void* d_ws, size_t ws_size,
                              hipStream_t stream) {
    const float* X = (const float*)d_in[0];
    const float* Y = (const float*)d_in[1];
    float* out = (float*)d_out;

    const int D = 1024;
    const int N = in_sizes[0] / D;   // 8192 rows of X
    const int M = in_sizes[1] / D;   // 8192 rows of Y

    float* x2 = (float*)d_ws;        // N floats
    float* y2 = x2 + N;              // M floats

    const size_t planeX = (size_t)N * D;
    const size_t planeY = (size_t)M * D;
    const size_t need = (size_t)(N + M) * sizeof(float)
                      + 2 * (planeX + planeY) * sizeof(__bf16);

    if (ws_size >= need) {
        // Fast path: precomputed bf16 hi/lo planes (L2-resident during GEMM).
        __bf16* Xh = (__bf16*)(y2 + M);
        __bf16* Xl = Xh + planeX;
        __bf16* Yh = Xl + planeX;
        __bf16* Yl = Yh + planeY;

        conv_split_norm<<<(N + 7) / 8, 256, 0, stream>>>(X, Xh, Xl, x2, N);
        conv_split_norm<<<(M + 7) / 8, 256, 0, stream>>>(Y, Yh, Yl, y2, M);

        dim3 grid(N / 128, M / 128);
        mgk_gemm_bf16<<<grid, 256, 0, stream>>>(Xh, Xl, Yh, Yl, x2, y2, out, M);
    } else {
        // Fallback: in-register split (no large workspace required).
        row_sqnorm<<<(N + 7) / 8, 256, 0, stream>>>(X, x2, N);
        row_sqnorm<<<(M + 7) / 8, 256, 0, stream>>>(Y, y2, M);

        dim3 grid(N / 128, M / 128);
        mgk_gemm_f32src<<<grid, 256, 0, stream>>>(X, Y, x2, y2, out, M);
    }
}
